// SimpleVisionTransformer_82532091560280
// MI455X (gfx1250) — matrix operand order repaired and checked by reading, not compiled, not run on hardware
//
#include <hip/hip_runtime.h>
#include <hip/hip_bf16.h>
#include <type_traits>

// ============================================================================
// ViT forward for MI455X (gfx1250), wave32 + WMMA bf16.
//
// All tensors that feed GEMMs are materialized as bf16 in memory (weights
// converted once per launch; LN/softmax/im2col emit bf16; GEMM epilogues can
// emit bf16). The GEMM therefore stages raw bf16 with the CDNA5 async
// global->LDS engine (global_load_async_to_lds_b128, ASYNCcnt) when exposed,
// falling back to b128 loads + LDS stores; fragments via ds_load_b128;
// v_wmma_f32_16x16x32_bf16 accumulation in fp32.
// Residual stream h and attention scores stay fp32.
// Block tile 128x128, K-step 32, double-buffered LDS (1 barrier/step).
// ============================================================================

typedef __attribute__((ext_vector_type(16))) __bf16 v16bf;
typedef __attribute__((ext_vector_type(8)))  float  v8f;
typedef int v4i_vs __attribute__((vector_size(16)));

#define BM 128
#define BN 128
#define BK 32
#define LDSU (BK / 2 + 4)        // LDS row stride in dwords (bf16 pairs): 20 -> 80B
#define TILEU ((BM + BN) * LDSU)

#if defined(__has_builtin)
#if __has_builtin(__builtin_amdgcn_global_load_async_to_lds_b128)
#define HAVE_ASYNC_LDS 1
#endif
#if __has_builtin(__builtin_amdgcn_s_wait_asynccnt)
#define HAVE_WAIT_ASYNC 1
#endif
#endif

__device__ __forceinline__ unsigned int bf16_bits(float f) {
  unsigned int u = __builtin_bit_cast(unsigned int, f);
  unsigned int r = u + 0x7FFFu + ((u >> 16) & 1u);   // round-to-nearest-even
  return r >> 16;
}

__device__ __forceinline__ long long zoff(int z, int dv, long long s1, long long s2) {
  return (long long)(z / dv) * s1 + (long long)(z % dv) * s2;
}

#ifdef HAVE_ASYNC_LDS
__device__ __forceinline__ void async_b128(const void* g, void* l) {
  __builtin_amdgcn_global_load_async_to_lds_b128(
      (__attribute__((address_space(1))) v4i_vs*)(g),
      (__attribute__((address_space(3))) v4i_vs*)(l), 0, 0);
}
__device__ __forceinline__ void wait_asynccnt0() {
#ifdef HAVE_WAIT_ASYNC
  __builtin_amdgcn_s_wait_asynccnt(0);
#else
  asm volatile("s_wait_asynccnt 0x0" ::: "memory");
#endif
}
#endif

// ---------------------------------------------------------------------------
// Guarded (edge) gather of one thread's 16-element share into packed dwords.
// ---------------------------------------------------------------------------
__device__ __forceinline__ void gather_guarded(
    unsigned int* __restrict__ u, const unsigned short* __restrict__ P,
    long long rstride, long long kstride, int row, int k0, int kk, int rmax, int K)
{
#pragma unroll
  for (int j = 0; j < 8; ++j) {
    const int ke0 = k0 + kk + 2 * j;
    unsigned int e0 = (row < rmax && ke0     < K) ? P[(long long)row * rstride + (long long)ke0 * kstride]       : 0u;
    unsigned int e1 = (row < rmax && ke0 + 1 < K) ? P[(long long)row * rstride + (long long)(ke0 + 1) * kstride] : 0u;
    u[j] = e0 | (e1 << 16);
  }
}

// ---------------------------------------------------------------------------
// Stage one K-step tile (A: BM x BK, B: BN x BK bf16) into an LDS buffer.
// Thread t covers row (t>>1), k-range (t&1)*16..+15 of both tiles.
// ---------------------------------------------------------------------------
template <bool WCONTIG>
__device__ __forceinline__ void stage_tile(
    unsigned int* __restrict__ dA, unsigned int* __restrict__ dB,
    const unsigned short* __restrict__ Ab, long long lda,
    const unsigned short* __restrict__ Wb, long long wrs, long long wcs,
    int m0, int n0, int k0, int M, int N, int K, int tid)
{
  const int r  = tid >> 1;
  const int kk = (tid & 1) * 16;
  const int gm = m0 + r;
  const int gn = n0 + r;
  const bool fullk = (k0 + BK <= K);
  unsigned int* la = dA + r * LDSU + (kk >> 1);
  unsigned int* lb = dB + r * LDSU + (kk >> 1);

  if (gm < M && fullk) {
    const unsigned short* g = Ab + (long long)gm * lda + (k0 + kk);
#ifdef HAVE_ASYNC_LDS
    async_b128(g, la);
    async_b128(g + 8, la + 4);
#else
    const uint4* s = (const uint4*)g;
    *(uint4*)(la)     = s[0];
    *(uint4*)(la + 4) = s[1];
#endif
  } else {
    unsigned int u[8];
    gather_guarded(u, Ab, lda, 1, gm, k0, kk, M, K);
    *(uint4*)(la)     = make_uint4(u[0], u[1], u[2], u[3]);
    *(uint4*)(la + 4) = make_uint4(u[4], u[5], u[6], u[7]);
  }

  if (WCONTIG && gn < N && fullk) {
    const unsigned short* g = Wb + (long long)gn * wrs + (k0 + kk);
#ifdef HAVE_ASYNC_LDS
    async_b128(g, lb);
    async_b128(g + 8, lb + 4);
#else
    const uint4* s = (const uint4*)g;
    *(uint4*)(lb)     = s[0];
    *(uint4*)(lb + 4) = s[1];
#endif
  } else {
    unsigned int u[8];
    gather_guarded(u, Wb, wrs, WCONTIG ? 1 : wcs, gn, k0, kk, N, K);
    *(uint4*)(lb)     = make_uint4(u[0], u[1], u[2], u[3]);
    *(uint4*)(lb + 4) = make_uint4(u[4], u[5], u[6], u[7]);
  }
}

// Assemble fragments from LDS and do the wave's 2x4 WMMAs.
__device__ __forceinline__ void compute_step(
    const unsigned int* __restrict__ bufA, const unsigned int* __restrict__ bufB,
    v8f (&acc)[2][4], int wm, int wn, int lmod, int lhi)
{
  v16bf bfrag[4];
#pragma unroll
  for (int tn = 0; tn < 4; ++tn) {
    const __bf16* rowp = (const __bf16*)(bufB + (wn + tn * 16 + lmod) * LDSU);
#pragma unroll
    for (int i = 0; i < 16; ++i)                 // contiguous K run per half-wave
      bfrag[tn][i] = rowp[16 * lhi + i];
  }
#pragma unroll
  for (int tm = 0; tm < 2; ++tm) {
    v16bf afrag;
    const __bf16* rowp = (const __bf16*)(bufA + (wm + tm * 16 + lmod) * LDSU);
#pragma unroll
    for (int i = 0; i < 16; ++i)                 // contiguous K run per half-wave
      afrag[i] = rowp[16 * lhi + i];
#pragma unroll
    for (int tn = 0; tn < 4; ++tn)
      acc[tm][tn] = __builtin_amdgcn_wmma_f32_16x16x32_bf16(
          false, afrag, false, bfrag[tn], (short)0, acc[tm][tn], false, false);
  }
}

// EPI: 0=bias 1=mask+scale 2=none 3=bias+gelu 4=bias+residual(fp32)
template <int EPI, bool WCONTIG, bool OUTBF>
__global__ __launch_bounds__(256)
void gemm_bf16_wmma(const unsigned short* __restrict__ A, long long lda, int adiv, long long as1, long long as2,
                    const unsigned short* __restrict__ W, long long wrs, long long wcs, int wdiv, long long ws1, long long ws2,
                    void* __restrict__ C, long long ldc, int cdiv, long long cs1, long long cs2,
                    const float* __restrict__ Cadd, const float* __restrict__ bias,
                    const float* __restrict__ mask, int mask_ld, float scale,
                    int M, int N, int K)
{
  __shared__ unsigned int ldsbuf[2][TILEU];

  const int z = blockIdx.z;
  const unsigned short* Ab = A + zoff(z, adiv, as1, as2);
  const unsigned short* Wb = W + zoff(z, wdiv, ws1, ws2);
  const long long coff = zoff(z, cdiv, cs1, cs2);
  float*          Cf = (float*)C + coff;
  unsigned short* Ch = (unsigned short*)C + coff;
  const float* Caddb = (EPI == 4) ? (Cadd + coff) : nullptr;

  const int m0 = blockIdx.y * BM;
  const int n0 = blockIdx.x * BN;
  const int tid  = threadIdx.x;
  const int lane = tid & 31;
  const int wid  = tid >> 5;
  const int wm = (wid >> 1) * 32;
  const int wn = (wid & 1) * 64;
  const int lmod = lane & 15;
  const int lhi  = lane >> 4;

  v8f acc[2][4] = {};
  const int ktiles = (K + BK - 1) / BK;

#ifdef HAVE_ASYNC_LDS
  // async copies land directly in LDS; they overlap the compute of the other buffer
  stage_tile<WCONTIG>(&ldsbuf[0][0], &ldsbuf[0][BM * LDSU], Ab, lda, Wb, wrs, wcs, m0, n0, 0, M, N, K, tid);
  wait_asynccnt0();
  __syncthreads();
  for (int kb = 0; ; ++kb) {
    const int p = kb & 1;
    const bool last = (kb + 1 == ktiles);
    if (!last)
      stage_tile<WCONTIG>(&ldsbuf[p ^ 1][0], &ldsbuf[p ^ 1][BM * LDSU], Ab, lda, Wb, wrs, wcs, m0, n0, (kb + 1) * BK, M, N, K, tid);
    compute_step(&ldsbuf[p][0], &ldsbuf[p][BM * LDSU], acc, wm, wn, lmod, lhi);
    if (last) break;
    wait_asynccnt0();
    __syncthreads();
  }
#else
  // register-prefetch fallback: load next tile to VGPRs while computing
  {
    const int r  = tid >> 1;
    const int kk = (tid & 1) * 16;
    unsigned int ua[8], ub[8];
    auto load_regs = [&](int k0) {
      const int gm = m0 + r, gn = n0 + r;
      const bool fullk = (k0 + BK <= K);
      if (gm < M && fullk) {
        const uint4* s = (const uint4*)(Ab + (long long)gm * lda + (k0 + kk));
        uint4 t0 = s[0], t1 = s[1];
        ua[0]=t0.x; ua[1]=t0.y; ua[2]=t0.z; ua[3]=t0.w; ua[4]=t1.x; ua[5]=t1.y; ua[6]=t1.z; ua[7]=t1.w;
      } else gather_guarded(ua, Ab, lda, 1, gm, k0, kk, M, K);
      if (WCONTIG && gn < N && fullk) {
        const uint4* s = (const uint4*)(Wb + (long long)gn * wrs + (k0 + kk));
        uint4 t0 = s[0], t1 = s[1];
        ub[0]=t0.x; ub[1]=t0.y; ub[2]=t0.z; ub[3]=t0.w; ub[4]=t1.x; ub[5]=t1.y; ub[6]=t1.z; ub[7]=t1.w;
      } else gather_guarded(ub, Wb, wrs, WCONTIG ? 1 : wcs, gn, k0, kk, N, K);
    };
    auto store_regs = [&](int p) {
      unsigned int* la = &ldsbuf[p][0] + r * LDSU + (kk >> 1);
      unsigned int* lb = &ldsbuf[p][BM * LDSU] + r * LDSU + (kk >> 1);
      *(uint4*)(la)     = make_uint4(ua[0], ua[1], ua[2], ua[3]);
      *(uint4*)(la + 4) = make_uint4(ua[4], ua[5], ua[6], ua[7]);
      *(uint4*)(lb)     = make_uint4(ub[0], ub[1], ub[2], ub[3]);
      *(uint4*)(lb + 4) = make_uint4(ub[4], ub[5], ub[6], ub[7]);
    };
    load_regs(0);
    store_regs(0);
    __syncthreads();
    for (int kb = 0; ; ++kb) {
      const int p = kb & 1;
      const bool last = (kb + 1 == ktiles);
      if (!last) load_regs((kb + 1) * BK);
      compute_step(&ldsbuf[p][0], &ldsbuf[p][BM * LDSU], acc, wm, wn, lmod, lhi);
      if (last) break;
      store_regs(p ^ 1);
      __syncthreads();
    }
  }
#endif

  // ---- epilogue (fast unguarded path for interior blocks) -----------------
  const bool inb = (m0 + BM <= M) && (n0 + BN <= N);
  auto epi = [&](auto guard_c) {
    constexpr bool GUARD = decltype(guard_c)::value;
#pragma unroll
    for (int tn = 0; tn < 4; ++tn) {
      const int n = n0 + wn + tn * 16 + lmod;
      float bv = 0.0f;
      if (EPI == 0 || EPI == 3 || EPI == 4) {
        if (!GUARD || n < N) bv = bias[n];
      }
#pragma unroll
      for (int tm = 0; tm < 2; ++tm) {
#pragma unroll
        for (int r = 0; r < 8; ++r) {
          const int m = m0 + wm + tm * 16 + r + 8 * lhi;
          if (!GUARD || (m < M && n < N)) {
            float v = acc[tm][tn][r] + bv;
            if (EPI == 1) v = (mask[(long long)m * mask_ld + n] > 0.0f) ? v * scale : -1e9f;
            if (EPI == 3) v = 0.5f * v * (1.0f + erff(v * 0.70710678118654752f));
            if (EPI == 4) v += Caddb[(long long)m * ldc + n];
            if (OUTBF) Ch[(long long)m * ldc + n] = (unsigned short)bf16_bits(v);
            else       Cf[(long long)m * ldc + n] = v;
          }
        }
      }
    }
  };
  if (inb) epi(std::false_type{});
  else     epi(std::true_type{});
}

// ============================================================================
// fp32 -> bf16 tensor conversion (weights, once per launch)
// ============================================================================
__global__ void vit_cvt_bf16(const float* __restrict__ in, unsigned short* __restrict__ out, long long n)
{
  long long i = (long long)blockIdx.x * blockDim.x + threadIdx.x;
  if (i < n) out[i] = (unsigned short)bf16_bits(in[i]);
}

// im2col: x[B,3,256,256] -> bf16 patches[B*256, 768], inner order (c,py,px)
__global__ void vit_im2col(const float* __restrict__ x, unsigned short* __restrict__ patches, long long total)
{
  long long i = (long long)blockIdx.x * blockDim.x + threadIdx.x;
  if (i >= total) return;
  int e = (int)(i % 768);
  long long t = i / 768;
  int p = (int)(t % 256);
  int b = (int)(t / 256);
  int c = e >> 8;
  int rem = e & 255;
  int py = rem >> 4, px = rem & 15;
  int gy = p >> 4,  gx = p & 15;
  patches[i] = (unsigned short)bf16_bits(x[(((long long)b * 3 + c) * 256 + (gy * 16 + py)) * 256 + (gx * 16 + px)]);
}

// h[b,s,d] = (s<256 ? tok[b,s,d] : 0) + pos_emb[s,d]   (fp32)
__global__ void vit_build_h(const float* __restrict__ tok, const float* __restrict__ pos,
                            float* __restrict__ h, long long total)
{
  long long i = (long long)blockIdx.x * blockDim.x + threadIdx.x;
  if (i >= total) return;
  int d = (int)(i % 768);
  long long t = i / 768;
  int s = (int)(t % 273);
  int b = (int)(t / 273);
  float v = (s < 256) ? tok[((long long)b * 256 + s) * 768 + d] : 0.0f;
  h[i] = v + pos[(long long)s * 768 + d];
}

// ============================================================================
// LayerNorm over 768 (fp32 in, bf16 out), one wave32 per row
// ============================================================================
__device__ __forceinline__ void ln_row_dev(const float* __restrict__ p, unsigned short* __restrict__ o,
                                           const float* __restrict__ g, const float* __restrict__ b,
                                           int lane)
{
  float vals[24];
  float s = 0.0f;
#pragma unroll
  for (int i = 0; i < 24; ++i) { vals[i] = p[lane + i * 32]; s += vals[i]; }
#pragma unroll
  for (int off = 16; off > 0; off >>= 1) s += __shfl_xor(s, off, 32);
  const float mean = s * (1.0f / 768.0f);
  float q = 0.0f;
#pragma unroll
  for (int i = 0; i < 24; ++i) { float d = vals[i] - mean; q += d * d; }
#pragma unroll
  for (int off = 16; off > 0; off >>= 1) q += __shfl_xor(q, off, 32);
  const float rstd = rsqrtf(q * (1.0f / 768.0f) + 1e-6f);
#pragma unroll
  for (int i = 0; i < 24; ++i) {
    const int d = lane + i * 32;
    o[d] = (unsigned short)bf16_bits((vals[i] - mean) * rstd * g[d] + b[d]);
  }
}

__global__ __launch_bounds__(256)
void vit_ln_rows(const float* __restrict__ in, unsigned short* __restrict__ out,
                 const float* __restrict__ g, const float* __restrict__ b, int rows)
{
  const int row = blockIdx.x * 8 + (threadIdx.x >> 5);
  if (row >= rows) return;
  ln_row_dev(in + (long long)row * 768, out + (long long)row * 768, g, b, threadIdx.x & 31);
}

__global__ __launch_bounds__(256)
void vit_ln_last(const float* __restrict__ h, unsigned short* __restrict__ out,
                 const float* __restrict__ g, const float* __restrict__ b, int Bn)
{
  const int row = blockIdx.x * 8 + (threadIdx.x >> 5);
  if (row >= Bn) return;
  ln_row_dev(h + ((long long)row * 273 + 272) * 768, out + (long long)row * 768, g, b, threadIdx.x & 31);
}

// ============================================================================
// Softmax: fp32 masked scores (len 273, stride 288) -> bf16 probs written
// in place over the row start (read-before-write within the owning wave).
// ============================================================================
__global__ __launch_bounds__(256)
void vit_softmax(float* __restrict__ sc, long long nrows)
{
  const long long row = (long long)blockIdx.x * 8 + (threadIdx.x >> 5);
  if (row >= nrows) return;
  const int lane = threadIdx.x & 31;
  float* p = sc + row * 288;
  unsigned short* pb = (unsigned short*)p;   // bf16 view of the same row
  float v[9];
  float mx = -1e30f;
#pragma unroll
  for (int i = 0; i < 9; ++i) {
    const int j = lane + i * 32;
    v[i] = (j < 273) ? p[j] : -1e30f;
    mx = fmaxf(mx, v[i]);
  }
#pragma unroll
  for (int off = 16; off > 0; off >>= 1) mx = fmaxf(mx, __shfl_xor(mx, off, 32));
  float s = 0.0f;
#pragma unroll
  for (int i = 0; i < 9; ++i) {
    const int j = lane + i * 32;
    if (j < 273) { v[i] = expf(v[i] - mx); s += v[i]; }
  }
#pragma unroll
  for (int off = 16; off > 0; off >>= 1) s += __shfl_xor(s, off, 32);
  const float inv = 1.0f / s;
#pragma unroll
  for (int i = 0; i < 9; ++i) {
    const int j = lane + i * 32;
    if (j < 273) pb[j] = (unsigned short)bf16_bits(v[i] * inv);
  }
}

// ============================================================================
// Host orchestration
// ============================================================================
static void launch_gemm(hipStream_t st,
                        const unsigned short* A, long long lda, int adiv, long long as1, long long as2,
                        const unsigned short* W, long long wrs, long long wcs, int wdiv, long long ws1, long long ws2,
                        void* C, long long ldc, int cdiv, long long cs1, long long cs2,
                        const float* Cadd, const float* bias,
                        const float* mask, int mask_ld, float scale,
                        int M, int N, int K, int epi, bool outbf, int Z)
{
  dim3 grid((N + BN - 1) / BN, (M + BM - 1) / BM, Z);
#define GEMM_ARGS A, lda, adiv, as1, as2, W, wrs, wcs, wdiv, ws1, ws2, \
                  C, ldc, cdiv, cs1, cs2, Cadd, bias, mask, mask_ld, scale, M, N, K
  if (epi == 2) { gemm_bf16_wmma<2, false, true><<<grid, 256, 0, st>>>(GEMM_ARGS); }
  else if (epi == 0 && !outbf) { gemm_bf16_wmma<0, true, false><<<grid, 256, 0, st>>>(GEMM_ARGS); }
  else if (epi == 0)           { gemm_bf16_wmma<0, true, true ><<<grid, 256, 0, st>>>(GEMM_ARGS); }
  else if (epi == 1)           { gemm_bf16_wmma<1, true, false><<<grid, 256, 0, st>>>(GEMM_ARGS); }
  else if (epi == 3)           { gemm_bf16_wmma<3, true, true ><<<grid, 256, 0, st>>>(GEMM_ARGS); }
  else                         { gemm_bf16_wmma<4, true, false><<<grid, 256, 0, st>>>(GEMM_ARGS); }
#undef GEMM_ARGS
}

static void launch_cvt(hipStream_t st, const float* in, unsigned short* out, long long n)
{
  vit_cvt_bf16<<<(unsigned)((n + 255) / 256), 256, 0, st>>>(in, out, n);
}

extern "C" void kernel_launch(void* const* d_in, const int* in_sizes, int n_in,
                              void* d_out, int out_size, void* d_ws, size_t ws_size,
                              hipStream_t stream)
{
  (void)in_sizes; (void)n_in; (void)out_size; (void)ws_size;
  const int Bn = 32, S = 273, D = 768, H = 12, L = 12, MD = 3072, NC = 100;
  const long long rows = (long long)Bn * S;   // 8736

  const float* x       = (const float*)d_in[0];
  const float* mask    = (const float*)d_in[1];
  const float* pos_emb = (const float*)d_in[2];
  const float* conv_w  = (const float*)d_in[3];
  const float* conv_b  = (const float*)d_in[4];
  const float* ln1_s   = (const float*)d_in[5];
  const float* ln1_b   = (const float*)d_in[6];
  const float* qkv_w   = (const float*)d_in[7];
  const float* qkv_b   = (const float*)d_in[8];
  const float* out_w   = (const float*)d_in[9];
  const float* out_b   = (const float*)d_in[10];
  const float* ln2_s   = (const float*)d_in[11];
  const float* ln2_b   = (const float*)d_in[12];
  const float* fc1_w   = (const float*)d_in[13];
  const float* fc1_b   = (const float*)d_in[14];
  const float* fc2_w   = (const float*)d_in[15];
  const float* fc2_b   = (const float*)d_in[16];
  const float* lnf_s   = (const float*)d_in[17];
  const float* lnf_b   = (const float*)d_in[18];
  const float* head_w  = (const float*)d_in[19];
  const float* head_b  = (const float*)d_in[20];
  float* out = (float*)d_out;

  // ---- workspace partition --------------------------------------------------
  char* base = (char*)d_ws;
  long long off = 0;
  auto alloc = [&](long long bytes) { void* p = base + off; off = (off + bytes + 255) & ~255LL; return p; };

  float*          h    = (float*)alloc(rows * D * 4);                 // fp32 residual
  unsigned short* zb   = (unsigned short*)alloc(rows * D * 2);        // LN out (bf16)
  unsigned short* qkvb = (unsigned short*)alloc(rows * 3 * D * 2);    // qkv (bf16)
  float*          scf  = (float*)alloc((long long)Bn * H * S * 288 * 4); // scores fp32 (+bf16 probs in place)
  unsigned short* obb  = (unsigned short*)alloc(rows * D * 2);        // attention out (bf16)
  unsigned short* z2b  = (unsigned short*)alloc(rows * (long long)MD * 2); // mlp hidden (bf16)
  unsigned short* hfb  = (unsigned short*)alloc((long long)Bn * D * 2);
  // bf16 weights
  const long long NCONV = (long long)D * D;
  const long long NQKV  = (long long)L * 3 * D * D;
  const long long NOUT  = (long long)L * D * D;
  const long long NFC1  = (long long)L * MD * D;
  const long long NFC2  = (long long)L * D * MD;
  const long long NHEAD = (long long)NC * D;
  unsigned short* wb_conv = (unsigned short*)alloc(NCONV * 2);
  unsigned short* wb_qkv  = (unsigned short*)alloc(NQKV * 2);
  unsigned short* wb_out  = (unsigned short*)alloc(NOUT * 2);
  unsigned short* wb_fc1  = (unsigned short*)alloc(NFC1 * 2);
  unsigned short* wb_fc2  = (unsigned short*)alloc(NFC2 * 2);
  unsigned short* wb_head = (unsigned short*)alloc(NHEAD * 2);
  unsigned short* patches = z2b;              // alias (used pre-layer only)
  float*          tok     = scf;              // alias (used pre-layer only)
  unsigned short* probs   = (unsigned short*)scf;  // bf16 view, row stride 576

  // ---- weight conversion (once per launch) ---------------------------------
  launch_cvt(stream, conv_w, wb_conv, NCONV);
  launch_cvt(stream, qkv_w,  wb_qkv,  NQKV);
  launch_cvt(stream, out_w,  wb_out,  NOUT);
  launch_cvt(stream, fc1_w,  wb_fc1,  NFC1);
  launch_cvt(stream, fc2_w,  wb_fc2,  NFC2);
  launch_cvt(stream, head_w, wb_head, NHEAD);

  // ---- patch embedding ------------------------------------------------------
  {
    long long tot = (long long)Bn * 256 * 768;
    vit_im2col<<<(unsigned)((tot + 255) / 256), 256, 0, stream>>>(x, patches, tot);
    launch_gemm(stream, patches, 768, 1, 0, 0, wb_conv, 768, 1, 1, 0, 0,
                tok, 768, 1, 0, 0, nullptr, conv_b, nullptr, 0, 0.0f,
                Bn * 256, D, 768, /*epi=*/0, /*outbf=*/false, 1);
    long long tot2 = rows * D;
    vit_build_h<<<(unsigned)((tot2 + 255) / 256), 256, 0, stream>>>(tok, pos_emb, h, tot2);
  }

  // ---- transformer layers ---------------------------------------------------
  for (int l = 0; l < L; ++l) {
    const unsigned short* qw = wb_qkv + (long long)l * 3 * D * D;
    const unsigned short* ow = wb_out + (long long)l * D * D;
    const unsigned short* f1w = wb_fc1 + (long long)l * MD * D;
    const unsigned short* f2w = wb_fc2 + (long long)l * D * MD;
    const float* qb  = qkv_b + (long long)l * 3 * D;
    const float* ob  = out_b + (long long)l * D;
    const float* f1b = fc1_b + (long long)l * MD;
    const float* f2b = fc2_b + (long long)l * D;

    vit_ln_rows<<<(unsigned)((rows + 7) / 8), 256, 0, stream>>>(h, zb, ln1_s + (long long)l * D, ln1_b + (long long)l * D, (int)rows);

    // qkv = LN1(h) @ qw^T + qb   (bf16 out)
    launch_gemm(stream, zb, D, 1, 0, 0, qw, D, 1, 1, 0, 0,
                qkvb, 3 * D, 1, 0, 0, nullptr, qb, nullptr, 0, 0.0f,
                (int)rows, 3 * D, D, 0, true, 1);

    // scores[b,h] = mask(q @ k^T * 0.125)  (fp32 out, stride 288)
    launch_gemm(stream,
                qkvb,     3 * D, H, (long long)S * 3 * D, 64,
                qkvb + D, 3 * D, 1, H, (long long)S * 3 * D, 64,
                scf, 288, 1, (long long)S * 288, 0,
                nullptr, nullptr, mask, S, 0.125f,
                S, S, 64, 1, false, Bn * H);

    vit_softmax<<<(unsigned)(((long long)Bn * H * S + 7) / 8), 256, 0, stream>>>(scf, (long long)Bn * H * S);

    // o[b,h] = P @ v   (P bf16 stride 576; v gathered K-major; bf16 out)
    launch_gemm(stream,
                probs, 576, 1, (long long)S * 576, 0,
                qkvb + 2 * D, 1, 3 * D, H, (long long)S * 3 * D, 64,
                obb, D, H, (long long)S * D, 64,
                nullptr, nullptr, nullptr, 0, 0.0f,
                S, 64, S, 2, true, Bn * H);

    // h += o @ ow^T + ob   (fp32 residual)
    launch_gemm(stream, obb, D, 1, 0, 0, ow, D, 1, 1, 0, 0,
                h, D, 1, 0, 0, h, ob, nullptr, 0, 0.0f,
                (int)rows, D, D, 4, false, 1);

    vit_ln_rows<<<(unsigned)((rows + 7) / 8), 256, 0, stream>>>(h, zb, ln2_s + (long long)l * D, ln2_b + (long long)l * D, (int)rows);

    // z2 = gelu(LN2(h) @ f1w^T + f1b)  (bf16 out)
    launch_gemm(stream, zb, D, 1, 0, 0, f1w, D, 1, 1, 0, 0,
                z2b, MD, 1, 0, 0, nullptr, f1b, nullptr, 0, 0.0f,
                (int)rows, MD, D, 3, true, 1);

    // h += z2 @ f2w^T + f2b  (fp32 residual)
    launch_gemm(stream, z2b, MD, 1, 0, 0, f2w, MD, 1, 1, 0, 0,
                h, D, 1, 0, 0, h, f2b, nullptr, 0, 0.0f,
                (int)rows, D, MD, 4, false, 1);
  }

  // ---- final LN on global token + head --------------------------------------
  vit_ln_last<<<4, 256, 0, stream>>>(h, hfb, lnf_s, lnf_b, Bn);
  launch_gemm(stream, hfb, D, 1, 0, 0, wb_head, D, 1, 1, 0, 0,
              out, NC, 1, 0, 0, nullptr, head_b, nullptr, 0, 0.0f,
              Bn, NC, D, 0, false, 1);
}
